// TransformerLayer_15427522527699
// MI455X (gfx1250) — matrix operand order repaired and checked by reading, not compiled, not run on hardware
//
#include <hip/hip_runtime.h>
#include <hip/hip_bf16.h>
#include <math.h>

// Problem constants (match reference)
#define NB   32      // batch
#define TT   1024    // sequence length
#define INKQ 64      // input_dim_kq (kq flat last-two dims: 2*32)
#define EKQ  128     // embed_dim_kq
#define NH   4       // heads
#define DK   32      // EKQ / NH
#define LOG2E 1.44269504088896340736f
// dv = 1 per head

typedef _Float16 v16h __attribute__((ext_vector_type(16)));
typedef _Float16 v8h  __attribute__((ext_vector_type(8)));
typedef float    v8f  __attribute__((ext_vector_type(8)));
typedef float    v4f  __attribute__((ext_vector_type(4)));

// Raw v_exp_f32 (base-2). Softmax args are <= 0; deep-negative inputs flush
// to ~0 which is the correct softmax weight. exp2(-inf) = 0.
__device__ __forceinline__ float fast_exp2(float x) {
    return __builtin_amdgcn_exp2f(x);
}

// ---------------------------------------------------------------------------
// Phase 1: K/Q/V projections.
//   K16    : [n][h][t][dk] f16
//   Q16    : [n][h][t][dk] f16, PRE-SCALED by log2(e) so WMMA scores are in
//            the base-2 domain and softmax uses raw v_exp_f32, no multiply.
//   Vf     : [n][h][t]     f32 (dv == 1)
// One block per (n,t): x row cached in LDS, 256 threads -> 128 K + 128 Q dots.
// ---------------------------------------------------------------------------
__global__ __launch_bounds__(256) void proj_kernel(
    const float* __restrict__ kq, const float* __restrict__ v,
    const float* __restrict__ Wk, const float* __restrict__ bk,
    const float* __restrict__ Wq, const float* __restrict__ bq,
    const float* __restrict__ Wv, const float* __restrict__ bv,
    _Float16* __restrict__ K16, _Float16* __restrict__ Q16,
    float* __restrict__ Vf)
{
    const int nt = blockIdx.x;            // 0 .. NB*TT-1
    const int n  = nt >> 10;
    const int t  = nt & (TT - 1);
    const int tid = threadIdx.x;

    __shared__ float xs[INKQ];
    // concat(kq[:,:,0], kq[:,:,1], axis=-1) is exactly the flat 64-float row
    const float* xrow = kq + (size_t)nt * INKQ;
    if (tid < INKQ) xs[tid] = xrow[tid];
    __syncthreads();

    const int e = tid & (EKQ - 1);
    const bool isK = (tid < EKQ);
    const float* W = isK ? Wk : Wq;
    const float* b = isK ? bk : bq;
    const float* wrow = W + (size_t)e * INKQ;
    float s = b[e];
#pragma unroll
    for (int i = 0; i < INKQ; ++i) s = fmaf(xs[i], wrow[i], s);

    const int h = e >> 5;                 // e / DK
    const int d = e & (DK - 1);
    const size_t idx = ((((size_t)n * NH + h) * TT) + t) * DK + d;
    if (isK) K16[idx] = (_Float16)s;
    else     Q16[idx] = (_Float16)(s * LOG2E);   // fold log2(e) into Q

    if (tid < NH) {                       // values: v[n,t,0]*Wv[h,0] + bv[h]
        float vv = v[nt];
        Vf[(((size_t)n * NH + tid) * TT) + t] = fmaf(vv, Wv[tid], bv[tid]);
    }
}

// ---------------------------------------------------------------------------
// Phase 2: flash attention. One wave32 per (n, h, 16-query tile).
// S^T tile = WMMA( A = K-tile (16x32 f16), B = Q-tile (32x16 f16) ) -> f32.
// C layout: lane = query column; key rows 0-7 live in lanes 0-15, rows 8-15
// in lanes 16-31. The two lane halves own DISJOINT key subsets, so each half
// runs its own online softmax with NO cross-lane traffic in the loop; the two
// (m, sum, acc) states are merged once at the end (flash combine).
// ---------------------------------------------------------------------------
__global__ __launch_bounds__(256) void attn_kernel(
    const _Float16* __restrict__ K16, const _Float16* __restrict__ Q16,
    const float* __restrict__ Vf, float* __restrict__ out)
{
    const int wave = (blockIdx.x * blockDim.x + threadIdx.x) >> 5; // 0..8191
    const int lane = threadIdx.x & 31;
    const int qt = wave & 63;             // query tile (T/16 = 64)
    const int nh = wave >> 6;             // n*NH + h
    const int n  = nh >> 2;
    const int h  = nh & 3;

    const _Float16* Ks = K16 + (size_t)nh * TT * DK;
    const _Float16* Qs = Q16 + (size_t)nh * TT * DK;
    const float*    Vs = Vf  + (size_t)nh * TT;

    const int lo = lane & 15;
    const int hi = lane >> 4;             // lane half (key rows 8*hi..8*hi+7)
    const int q0 = qt * 16;

    // B operand (Q tile): lane holds column q0+lo; interleaved layout:
    // halves 0..7 -> dk[8*hi .. 8*hi+7], halves 8..15 -> dk[+16 .. +23]
    v16h bq;
    {
        const _Float16* p = Qs + (size_t)(q0 + lo) * DK + 8 * hi;
        v8h a0 = *(const v8h*)(p);
        v8h a1 = *(const v8h*)(p + 16);
#pragma unroll
        for (int i = 0; i < 8; ++i) { bq[i] = a0[i]; bq[i + 8] = a1[i]; }
    }

    // Per-lane-half online softmax state (base-2 domain)
    float mrun = -INFINITY, ssum = 0.0f, acc = 0.0f;

    for (int kt = 0; kt <= qt; ++kt) {
        const int k0 = kt * 16;
        // A operand (K tile): lane holds row k0+lo; interleaved K layout:
        // halves 0..7 -> dk[8*hi .. 8*hi+7], halves 8..15 -> dk[+16 .. +23]
        v16h aK;
        {
            const _Float16* p = Ks + (size_t)(k0 + lo) * DK + 8 * hi;
            v8h a0 = *(const v8h*)(p);
            v8h a1 = *(const v8h*)(p + 16);
#pragma unroll
            for (int i = 0; i < 8; ++i) { aK[i] = a0[i]; aK[i + 8] = a1[i]; }
        }

        v8f c = {};
        c = __builtin_amdgcn_wmma_f32_16x16x32_f16(
                false, aK, false, bq, (short)0, c, false, false);

        // c[r] = log2-domain score( key = k0 + r + 8*hi , query = q0 + lo )
        float s[8];
#pragma unroll
        for (int r = 0; r < 8; ++r) s[r] = c[r];

        if (kt == qt) {                   // only the diagonal tile needs a mask
#pragma unroll
            for (int r = 0; r < 8; ++r) {
                int m = r + 8 * hi;
                bool allow = (m < lo) || (k0 == 0 && m == 0 && lo == 0);
                if (!allow) s[r] = -INFINITY;
            }
        }

        float tm = s[0];
#pragma unroll
        for (int r = 1; r < 8; ++r) tm = fmaxf(tm, s[r]);

        float nm  = fmaxf(mrun, tm);
        float nms = (nm == -INFINITY) ? 0.0f : nm; // NaN guard
        float scale = fast_exp2(mrun - nms);       // mrun=-inf -> 0

        v4f v0 = *(const v4f*)(Vs + k0 + 8 * hi);
        v4f v1 = *(const v4f*)(Vs + k0 + 8 * hi + 4);

        float ps = 0.0f, pv = 0.0f;
#pragma unroll
        for (int r = 0; r < 8; ++r) {
            float p = fast_exp2(s[r] - nms);       // s=-inf -> 0
            ps += p;
            float vv = (r < 4) ? v0[r] : v1[r - 4];
            pv = fmaf(p, vv, pv);
        }

        ssum = ssum * scale + ps;
        acc  = acc  * scale + pv;
        mrun = nm;
    }

    // Merge the two lane-half states (disjoint key sets) once.
    float mo = __shfl_xor(mrun, 16, 32);
    float so = __shfl_xor(ssum, 16, 32);
    float ao = __shfl_xor(acc,  16, 32);
    float M  = fmaxf(mrun, mo);
    float Ms = (M == -INFINITY) ? 0.0f : M;
    float w0 = fast_exp2(mrun - Ms);
    float w1 = fast_exp2(mo   - Ms);
    float S  = ssum * w0 + so * w1;
    float A  = acc  * w0 + ao * w1;

    if (hi == 0) {                         // out[n][q0+lo][h], dv = 1
        out[((size_t)n * TT + q0 + lo) * NH + h] = A / S;
    }
}

// ---------------------------------------------------------------------------
extern "C" void kernel_launch(void* const* d_in, const int* in_sizes, int n_in,
                              void* d_out, int out_size, void* d_ws, size_t ws_size,
                              hipStream_t stream)
{
    const float* kq = (const float*)d_in[0];
    const float* v  = (const float*)d_in[1];
    const float* Wk = (const float*)d_in[2];
    const float* bk = (const float*)d_in[3];
    const float* Wq = (const float*)d_in[4];
    const float* bq = (const float*)d_in[5];
    const float* Wv = (const float*)d_in[6];
    const float* bv = (const float*)d_in[7];
    float* out = (float*)d_out;

    // Workspace layout: 8 MiB K16 | 8 MiB Q16 | 512 KiB Vf
    char* ws = (char*)d_ws;
    _Float16* K16 = (_Float16*)(ws);
    _Float16* Q16 = (_Float16*)(ws + (size_t)NB * NH * TT * DK * sizeof(_Float16));
    float*    Vf  = (float*)   (ws + (size_t)2 * NB * NH * TT * DK * sizeof(_Float16));

    proj_kernel<<<NB * TT, 256, 0, stream>>>(kq, v, Wk, bk, Wq, bq, Wv, bv,
                                             K16, Q16, Vf);

    // NB*NH*(TT/16) = 8192 waves, 8 waves (256 threads) per block
    const int total_waves = NB * NH * (TT / 16);
    attn_kernel<<<total_waves * 32 / 256, 256, 0, stream>>>(K16, Q16, Vf, out);
}
